// Model_17154099381014
// MI455X (gfx1250) — matrix operand order repaired and checked by reading, not compiled, not run on hardware
//
#include <hip/hip_runtime.h>
#include <hip/hip_bf16.h>
#include <math.h>

typedef __attribute__((ext_vector_type(16))) __bf16 v16bf;
typedef __attribute__((ext_vector_type(8)))  __bf16 v8bf;
typedef __attribute__((ext_vector_type(4)))  __bf16 v4bf;
typedef __attribute__((ext_vector_type(8)))  float  v8f;

#define NTOK 4096
#define DDIM 1024
#define FDIM 4096
#define NEXP 8
#define MAXPAIR (2*NTOK)

#define BM 128
#define BN 256
#define BK 32
#define LDP (BK+8)   // padded LDS row stride (80B: 16B-aligned, conflict-friendly)

__device__ __forceinline__ float gelu_exact(float x){
    return 0.5f * x * (1.0f + erff(x * 0.70710678118654752440f));
}

__global__ void k_zero_cnt(int* cnt){
    if (threadIdx.x < NEXP) cnt[threadIdx.x] = 0;
}

__global__ void k_cvt_x(const float* __restrict__ x, __bf16* __restrict__ xb){
    int i = (blockIdx.x*blockDim.x + threadIdx.x)*4;
    float4 v = *(const float4*)(x + i);
    v4bf o;
    o[0] = (__bf16)v.x; o[1] = (__bf16)v.y; o[2] = (__bf16)v.z; o[3] = (__bf16)v.w;
    *(v4bf*)(xb + i) = o;
}

// softmax -> mask -> top2 -> renorm -> route (atomic slot per expert)
__global__ void k_gates(const float* __restrict__ logits, const int* __restrict__ masks,
                        int* __restrict__ cnt, int* __restrict__ pairE,
                        int* __restrict__ pairSlot, float* __restrict__ pairGate){
    int b = blockIdx.x*blockDim.x + threadIdx.x;
    if (b >= NTOK) return;
    float l[NEXP]; float mx = -1e30f;
    #pragma unroll
    for (int e=0;e<NEXP;e++){ l[e] = logits[b*NEXP+e]; mx = fmaxf(mx, l[e]); }
    float g[NEXP]; float s = 0.f;
    #pragma unroll
    for (int e=0;e<NEXP;e++){ g[e] = __expf(l[e]-mx); s += g[e]; }
    float inv = 1.f/s;
    #pragma unroll
    for (int e=0;e<NEXP;e++){ g[e] *= inv; if (masks[b*NEXP+e] != 1) g[e] = 0.f; }
    int i1 = 0; float v1 = -1.f;
    #pragma unroll
    for (int e=0;e<NEXP;e++){ if (g[e] > v1){ v1 = g[e]; i1 = e; } }
    int i2 = 0; float v2 = -1.f;
    #pragma unroll
    for (int e=0;e<NEXP;e++){ if (e != i1 && g[e] > v2){ v2 = g[e]; i2 = e; } }
    float denom = v1 + v2 + 1e-9f;
    float g1 = v1/denom, g2 = v2/denom;
    int e0=-1,s0=0,e1=-1,s1=0; float q0=0.f,q1=0.f;
    if (g1 > 0.f){ e0 = i1; s0 = atomicAdd(&cnt[i1],1); q0 = g1; }
    if (g2 > 0.f){ e1 = i2; s1 = atomicAdd(&cnt[i2],1); q1 = g2; }
    pairE[b*2+0]=e0; pairSlot[b*2+0]=s0; pairGate[b*2+0]=q0;
    pairE[b*2+1]=e1; pairSlot[b*2+1]=s1; pairGate[b*2+1]=q1;
}

__global__ void k_prefix(const int* __restrict__ cnt, int* __restrict__ off){
    if (threadIdx.x == 0){
        int r = 0;
        for (int e=0;e<NEXP;e++){ off[e] = r; r += cnt[e]; }
    }
}

__global__ void k_fill(const int* __restrict__ pairE, const int* __restrict__ pairSlot,
                       const float* __restrict__ pairGate, const int* __restrict__ off,
                       int* __restrict__ rowTok, float* __restrict__ rowGate){
    int b = blockIdx.x*blockDim.x + threadIdx.x;
    if (b >= NTOK) return;
    #pragma unroll
    for (int j=0;j<2;j++){
        int e = pairE[b*2+j];
        if (e >= 0){
            int g = off[e] + pairSlot[b*2+j];
            rowTok[g] = b; rowGate[g] = pairGate[b*2+j];
        }
    }
}

__device__ __forceinline__ v16bf join16(v8bf lo, v8bf hi){
    return __builtin_shufflevector(lo, hi, 0,1,2,3,4,5,6,7,8,9,10,11,12,13,14,15);
}

// Tiled bf16 WMMA GEMM over the compacted rows of one expert.
// Double-buffered LDS: next tile's global loads are issued before the barrier,
// current buffer is computed, then staged registers are converted+stored into
// the alternate buffer. One __syncthreads per K-step.
// GATHER: A rows indirected through rowTok (pass A: A = Xbf16).
// DO_GELU: epilogue GELU (pass A). DO_BIAS: epilogue (+b2)*gate (pass B).
template<bool GATHER, bool DO_GELU, bool DO_BIAS>
__global__ __launch_bounds__(256) void k_gemm(
    const __bf16* __restrict__ A, const float* __restrict__ W,
    const float* __restrict__ b2, const int* __restrict__ rowTok,
    const float* __restrict__ rowGate, const int* __restrict__ cnt,
    const int* __restrict__ off, __bf16* __restrict__ Out,
    int K, int N, int ldA)
{
    const int e = blockIdx.z;
    const int cntE = cnt[e];
    const int tileM0 = blockIdx.x * BM;
    if (tileM0 >= cntE) return;
    const int tileN0 = blockIdx.y * BN;
    const long rowBase = (long)off[e] + tileM0;
    const float* We = W + (size_t)e * (size_t)K * (size_t)N;

    __shared__ __bf16 As[2][BM][LDP];
    __shared__ __bf16 Bs[2][BN][LDP];

    const int tid  = threadIdx.x;
    const int lane = tid & 31;
    const int wave = tid >> 5;
    const int l16  = lane & 15;
    const int lh   = lane >> 4;
    const int wm   = (wave & 1) * 64;   // 2 waves over M
    const int wn   = (wave >> 1) * 64;  // 4 waves over N

    v8f acc[4][4];
    #pragma unroll
    for (int i=0;i<4;i++)
        #pragma unroll
        for (int j=0;j<4;j++) acc[i][j] = (v8f)(0.0f);

    // ---- K-invariant staging addresses ----
    const int ar = tid >> 1;  // 0..127: A row within tile
    const int ah = tid & 1;   // which 16-element half of the 32 K values
    bool avalid = (tileM0 + ar) < cntE;
    long srcRow;
    if (GATHER) srcRow = avalid ? (long)rowTok[rowBase + ar] : 0l;
    else        srcRow = avalid ? (rowBase + ar) : 0l;
    const __bf16* aSrcBase = A + srcRow*(long)ldA + ah*16;
    const float*  wSrcBase = We + (size_t)(tileN0 + tid);

    // ---- register staging for the next tile ----
    v8bf ax0, ax1;
    float wreg[BK];

    auto issueLoads = [&](int kb){
        const v8bf* s = (const v8bf*)(aSrcBase + kb);
        ax0 = s[0]; ax1 = s[1];
        const float* wp = wSrcBase + (size_t)kb * (size_t)N;
        #pragma unroll
        for (int k=0;k<BK;k++) wreg[k] = wp[(size_t)k * (size_t)N];
    };
    auto storeTile = [&](int bsel){
        *(v8bf*)&As[bsel][ar][ah*16 + 0] = ax0;
        *(v8bf*)&As[bsel][ar][ah*16 + 8] = ax1;
        __align__(16) __bf16 tmp[BK];
        #pragma unroll
        for (int k=0;k<BK;k++) tmp[k] = (__bf16)wreg[k];
        #pragma unroll
        for (int k=0;k<BK;k+=8) *(v8bf*)&Bs[bsel][tid][k] = *(const v8bf*)&tmp[k];
    };

    issueLoads(0);
    storeTile(0);

    int buf = 0;
    for (int kb = 0; kb < K; kb += BK){
        const bool more = (kb + BK) < K;
        if (more) issueLoads(kb + BK);   // prefetch next tile into registers
        __syncthreads();                 // buffer `buf` fully staged

        // A fragment (16x32 MxK): lanes 0-15 row M=l16 hold K 0..7 & 16..23,
        // lanes 16-31 hold K 8..15 & 24..31.
        v16bf af[4], bfr[4];
        #pragma unroll
        for (int tm=0;tm<4;tm++){
            const __bf16* ap = &As[buf][wm + tm*16 + l16][lh*8];
            af[tm] = join16(*(const v8bf*)ap, *(const v8bf*)(ap + 16));
        }
        // B fragment (32x16 KxN): lane = column N=l16; half lh holds K lh*8..+7 & 16+lh*8..+7
        #pragma unroll
        for (int tn=0;tn<4;tn++){
            const __bf16* bp = &Bs[buf][wn + tn*16 + l16][lh*8];
            bfr[tn] = join16(*(const v8bf*)bp, *(const v8bf*)(bp + 16));
        }
        #pragma unroll
        for (int tm=0;tm<4;tm++)
            #pragma unroll
            for (int tn=0;tn<4;tn++)
                acc[tm][tn] = __builtin_amdgcn_wmma_f32_16x16x32_bf16(
                    false, af[tm], false, bfr[tn], (short)0, acc[tm][tn], false, false);

        if (more){
            storeTile(buf ^ 1);          // safe: buf^1 last read before this iter's sync
            buf ^= 1;
        }
    }

    // Epilogue. C/D layout: VGPR r, lanes 0-15 -> M=r, lanes 16-31 -> M=r+8; N=l16.
    #pragma unroll
    for (int tm=0;tm<4;tm++){
        const int mBase = wm + tm*16 + lh*8;
        #pragma unroll
        for (int tn=0;tn<4;tn++){
            const int n = tileN0 + wn + tn*16 + l16;
            v8f c = acc[tm][tn];
            #pragma unroll
            for (int r=0;r<8;r++){
                int m = mBase + r;
                if (tileM0 + m < cntE){
                    long row = rowBase + m;
                    float v = c[r];
                    if (DO_GELU) v = gelu_exact(v);
                    if (DO_BIAS) v = (v + b2[(size_t)e*(size_t)N + n]) * rowGate[row];
                    Out[row*(long)N + n] = (__bf16)v;
                }
            }
        }
    }
}

__global__ void k_combine(const __bf16* __restrict__ Y, const int* __restrict__ pairE,
                          const int* __restrict__ pairSlot, const int* __restrict__ off,
                          __hip_bfloat16* __restrict__ out){
    int b = blockIdx.x;
    int e0 = pairE[b*2+0], e1 = pairE[b*2+1];
    long r0 = (e0 >= 0) ? (long)(off[e0] + pairSlot[b*2+0]) : -1l;
    long r1 = (e1 >= 0) ? (long)(off[e1] + pairSlot[b*2+1]) : -1l;
    #pragma unroll
    for (int i=0;i<4;i++){
        int d = threadIdx.x + i*256;
        float acc = 0.f;
        if (r0 >= 0) acc += (float)Y[r0*DDIM + d];
        if (r1 >= 0) acc += (float)Y[r1*DDIM + d];
        out[(long)b*DDIM + d] = __float2bfloat16(acc);
    }
}

extern "C" void kernel_launch(void* const* d_in, const int* in_sizes, int n_in,
                              void* d_out, int out_size, void* d_ws, size_t ws_size,
                              hipStream_t stream){
    const float* x      = (const float*)d_in[0];  // [B,D]
    const float* logits = (const float*)d_in[1];  // [B,E]
    const int*   masks  = (const int*)  d_in[2];  // [B,E]
    const float* W1     = (const float*)d_in[3];  // [E,D,F]
    const float* W2     = (const float*)d_in[4];  // [E,F,D]
    const float* b2     = (const float*)d_in[5];  // [E,D]
    __hip_bfloat16* out = (__hip_bfloat16*)d_out; // [B,D]

    char* w = (char*)d_ws;
    size_t o = 0;
    auto take = [&](size_t bytes)->char*{
        char* p = w + o;
        o = (o + bytes + 255) & ~(size_t)255;
        return p;
    };
    __bf16* Xb      = (__bf16*)take((size_t)NTOK*DDIM*2);     // 8 MB
    __bf16* H       = (__bf16*)take((size_t)MAXPAIR*FDIM*2);  // 64 MB
    __bf16* Y       = (__bf16*)take((size_t)MAXPAIR*DDIM*2);  // 16 MB
    int*    cnt     = (int*)  take(NEXP*sizeof(int));
    int*    off     = (int*)  take(NEXP*sizeof(int));
    int*    pairE   = (int*)  take(MAXPAIR*sizeof(int));
    int*    pairSlot= (int*)  take(MAXPAIR*sizeof(int));
    float*  pairGate= (float*)take(MAXPAIR*sizeof(float));
    int*    rowTok  = (int*)  take(MAXPAIR*sizeof(int));
    float*  rowGate = (float*)take(MAXPAIR*sizeof(float));

    k_zero_cnt<<<1, NEXP, 0, stream>>>(cnt);
    k_cvt_x<<<(NTOK*DDIM/4)/256, 256, 0, stream>>>(x, Xb);
    k_gates<<<NTOK/256, 256, 0, stream>>>(logits, masks, cnt, pairE, pairSlot, pairGate);
    k_prefix<<<1, 1, 0, stream>>>(cnt, off);
    k_fill<<<NTOK/256, 256, 0, stream>>>(pairE, pairSlot, pairGate, off, rowTok, rowGate);

    // Pass A: H = GELU(X @ W1_e), K=D, N=F, gathered A rows
    dim3 gA(NTOK/BM, FDIM/BN, NEXP);
    k_gemm<true, true, false><<<gA, 256, 0, stream>>>(
        Xb, W1, nullptr, rowTok, rowGate, cnt, off, H, DDIM, FDIM, DDIM);

    // Pass B: Y = gate * (H @ W2_e + b2_e), K=F, N=D
    dim3 gB(NTOK/BM, DDIM/BN, NEXP);
    k_gemm<false, false, true><<<gB, 256, 0, stream>>>(
        H, W2, b2, rowTok, rowGate, cnt, off, Y, FDIM, DDIM, FDIM);

    k_combine<<<NTOK, 256, 0, stream>>>(Y, pairE, pairSlot, off, out);
}
